// MultipleExpertsLinear_85469849191072
// MI455X (gfx1250) — matrix operand order repaired and checked by reading, not compiled, not run on hardware
//
#include <hip/hip_runtime.h>

typedef __attribute__((ext_vector_type(16))) __bf16 v16bf;
typedef __attribute__((ext_vector_type(8)))  __bf16 v8bf;
typedef __attribute__((ext_vector_type(4)))  __bf16 v4bf;
typedef __attribute__((ext_vector_type(8)))  float  v8f;
typedef __attribute__((ext_vector_type(4)))  float  v4f;
typedef __attribute__((ext_vector_type(4)))  int    v4i;

namespace {
constexpr int kN   = 16384;
constexpr int kIn  = 512;
constexpr int kOut = 512;
constexpr int kE   = 8;
constexpr int BM = 128, BN = 128, BK = 32;
constexpr int LDA = BK + 4;                    // fp32 tile row stride (fallback kernel)
constexpr int LDB = BK + 4;
constexpr int LDT = BK + 8;                    // 40 bf16: conflict-free ds_load_b128
constexpr int STEPS_PER_EXPERT = kIn / BK;     // 16
constexpr int KSTEPS = (kE * kIn) / BK;        // 128
constexpr size_t kWElems = (size_t)kE * kOut * kIn;  // 2M weight elements
}

#if __has_builtin(__builtin_amdgcn_global_load_async_to_lds_b128)
#define MOE_ASYNC 1
#else
#define MOE_ASYNC 0
#endif

#if MOE_ASYNC
typedef __attribute__((address_space(1))) v4i as1_v4i;
typedef __attribute__((address_space(3))) v4i as3_v4i;
__device__ __forceinline__ as3_v4i* moe_lds(void* p) {
  return (as3_v4i*)(unsigned int)(unsigned long long)p;   // low 32 bits = LDS offset
}
__device__ __forceinline__ as1_v4i* moe_glb(const void* p) {
  return (as1_v4i*)(unsigned long long)p;
}
__device__ __forceinline__ void moe_wait_async() {
#if __has_builtin(__builtin_amdgcn_s_wait_asynccnt)
  __builtin_amdgcn_s_wait_asynccnt(0);
#else
  asm volatile("s_wait_asynccnt 0x0" ::: "memory");
#endif
}
#endif

union V16 { v16bf v; v8bf h[2]; };

// ---------------- prepack: weight fp32 -> bf16 hi/lo (once per launch) --------
__global__ __launch_bounds__(256) void moe_prepack_w(const float* __restrict__ w,
                                                     __bf16* __restrict__ whi,
                                                     __bf16* __restrict__ wlo) {
  const size_t idx4 = ((size_t)blockIdx.x * 256 + threadIdx.x) * 4;
  v4f v = *(const v4f*)(w + idx4);
  v4bf hi, lo;
#pragma unroll
  for (int i = 0; i < 4; ++i) {
    float f  = v[i];
    __bf16 h = (__bf16)f;
    hi[i] = h;
    lo[i] = (__bf16)(f - (float)h);
  }
  *(v4bf*)(whi + idx4) = hi;
  *(v4bf*)(wlo + idx4) = lo;
}

// ---------------- main GEMM: bf16x3, prepacked weights ------------------------
__global__ __launch_bounds__(256) void moe_gemm_main(
    const float*  __restrict__ x,     // (N, IN)
    const float*  __restrict__ coef,  // (N, E)
    const __bf16* __restrict__ whi,   // (E, OUT, IN) bf16 hi
    const __bf16* __restrict__ wlo,   // (E, OUT, IN) bf16 lo
    const float*  __restrict__ bias,  // (E, OUT)
    float*        __restrict__ out) { // (N, OUT)
  __shared__ __align__(16) __bf16 lAhi[BM * LDT];
  __shared__ __align__(16) __bf16 lAlo[BM * LDT];
  __shared__ __align__(16) __bf16 lBhi[BN * LDT];
  __shared__ __align__(16) __bf16 lBlo[BN * LDT];
  __shared__ __align__(16) float  lC[BM * kE];

  const int tid   = threadIdx.x;
  const int lane  = tid & 31;
  const int wave  = tid >> 5;
  const int wm    = wave >> 2;        // 0..1 : 64-row slab
  const int wn    = wave & 3;         // 0..3 : 32-col slab
  const int lhalf = lane >> 4;
  const int l16   = lane & 15;

  const int colBase = blockIdx.x * BN;
  const int rowBase = blockIdx.y * BM;

  // coefficients tile: 128 x 8 floats, one b128 per thread
  ((v4f*)lC)[tid] = ((const v4f*)(coef + (size_t)rowBase * kE))[tid];
  __syncthreads();   // lC read during A staging below

  v8f acc[4][2];
#pragma unroll
  for (int t = 0; t < 4; ++t)
#pragma unroll
    for (int u = 0; u < 2; ++u) acc[t][u] = v8f{};

  const int ak0 = lhalf * 16;   // A frag: K ak0..ak0+15
  const int bk0 = lhalf * 16;   // B frag: K bk0..bk0+15

  for (int kk = 0; kk < KSTEPS; ++kk) {
    const int e   = kk / STEPS_PER_EXPERT;
    const int kin = (kk % STEPS_PER_EXPERT) * BK;

    // ---- B: stage prepacked bf16 hi/lo tiles (no VALU work) -----------------
    {
      const __bf16* gBh = whi + ((size_t)e * kOut + colBase) * kIn + kin;
      const __bf16* gBl = wlo + ((size_t)e * kOut + colBase) * kIn + kin;
#pragma unroll
      for (int j = 0; j < 4; ++j) {
        const int cc = tid + (j & 1) * 256;      // 0..511: chunk within buffer
        const int r  = cc >> 2;                  // out-col row 0..127
        const int q  = (cc & 3) * 8;             // bf16 offset of 16B chunk
        const __bf16* src = (j < 2) ? (gBh + (size_t)r * kIn + q)
                                    : (gBl + (size_t)r * kIn + q);
        __bf16* dst = (j < 2) ? &lBhi[r * LDT + q] : &lBlo[r * LDT + q];
#if MOE_ASYNC
        __builtin_amdgcn_global_load_async_to_lds_b128(moe_glb(src), moe_lds(dst), 0, 0);
#else
        *(v4i*)dst = *(const v4i*)src;
#endif
      }
    }

    // ---- A: global->reg, coef-scale + bf16 split once per tile, -> LDS ------
    {
      const float* gA = x + (size_t)rowBase * kIn + kin;
#pragma unroll
      for (int j = 0; j < 4; ++j) {
        const int c = tid + j * 256;   // 0..1023
        const int r = c >> 3;          // row 0..127
        const int q = (c & 7) * 4;     // float offset of 16B chunk
        v4f v = *(const v4f*)(gA + (size_t)r * kIn + q);
        const float cf = lC[r * kE + e];
        v4bf hi, lo;
#pragma unroll
        for (int i = 0; i < 4; ++i) {
          float f  = v[i] * cf;
          __bf16 h = (__bf16)f;
          hi[i] = h;
          lo[i] = (__bf16)(f - (float)h);
        }
        *(v4bf*)&lAhi[r * LDT + q] = hi;
        *(v4bf*)&lAlo[r * LDT + q] = lo;
      }
    }
#if MOE_ASYNC
    moe_wait_async();
#endif
    __syncthreads();

    // ---- fragments straight from bf16 LDS -----------------------------------
    V16 bhi[2], blo[2];
#pragma unroll
    for (int u = 0; u < 2; ++u) {
      const int brow = wn * 32 + u * 16 + l16;
      bhi[u].h[0] = *(const v8bf*)&lBhi[brow * LDT + bk0];
      bhi[u].h[1] = *(const v8bf*)&lBhi[brow * LDT + bk0 + 8];
      blo[u].h[0] = *(const v8bf*)&lBlo[brow * LDT + bk0];
      blo[u].h[1] = *(const v8bf*)&lBlo[brow * LDT + bk0 + 8];
    }
#pragma unroll
    for (int t = 0; t < 4; ++t) {
      const int arow = wm * 64 + t * 16 + l16;
      V16 ahi, alo;
      ahi.h[0] = *(const v8bf*)&lAhi[arow * LDT + ak0];
      ahi.h[1] = *(const v8bf*)&lAhi[arow * LDT + ak0 + 8];
      alo.h[0] = *(const v8bf*)&lAlo[arow * LDT + ak0];
      alo.h[1] = *(const v8bf*)&lAlo[arow * LDT + ak0 + 8];
#pragma unroll
      for (int u = 0; u < 2; ++u) {
        acc[t][u] = __builtin_amdgcn_wmma_f32_16x16x32_bf16(
            false, ahi.v, false, bhi[u].v, (short)0, acc[t][u], false, false);
        acc[t][u] = __builtin_amdgcn_wmma_f32_16x16x32_bf16(
            false, ahi.v, false, blo[u].v, (short)0, acc[t][u], false, false);
        acc[t][u] = __builtin_amdgcn_wmma_f32_16x16x32_bf16(
            false, alo.v, false, bhi[u].v, (short)0, acc[t][u], false, false);
      }
    }
    __syncthreads();
  }

  // ---- epilogue: out += coef @ bias ; C layout: M = j + 8*lhalf, N = l16 ----
#pragma unroll
  for (int u = 0; u < 2; ++u) {
    const int col = colBase + wn * 32 + u * 16 + l16;
    float bc[8];
#pragma unroll
    for (int q = 0; q < 8; ++q) bc[q] = bias[q * kOut + col];
#pragma unroll
    for (int t = 0; t < 4; ++t) {
#pragma unroll
      for (int j = 0; j < 8; ++j) {
        const int lrow = wm * 64 + t * 16 + lhalf * 8 + j;
        float bd = 0.f;
#pragma unroll
        for (int q = 0; q < 8; ++q) bd += lC[lrow * kE + q] * bc[q];
        out[(size_t)(rowBase + lrow) * kOut + col] = acc[t][u][j] + bd;
      }
    }
  }
}

// ---------------- fallback (no workspace): round-2 all-in-one kernel ---------
__device__ __forceinline__ void moe_split16(const float* f, v16bf& hi, v16bf& lo) {
#pragma unroll
  for (int i = 0; i < 16; ++i) {
    float v  = f[i];
    __bf16 h = (__bf16)v;
    hi[i] = h;
    lo[i] = (__bf16)(v - (float)h);
  }
}

__global__ __launch_bounds__(256) void moe_gemm_fallback(
    const float* __restrict__ x, const float* __restrict__ coef,
    const float* __restrict__ w, const float* __restrict__ bias,
    float* __restrict__ out) {
  __shared__ float lA[BM * LDA];
  __shared__ float lB[BN * LDB];
  __shared__ float lC[BM * kE];

  const int tid = threadIdx.x, lane = tid & 31, wave = tid >> 5;
  const int wm = wave >> 2, wn = wave & 3, lhalf = lane >> 4, l16 = lane & 15;
  const int colBase = blockIdx.x * BN, rowBase = blockIdx.y * BM;

  ((v4f*)lC)[tid] = ((const v4f*)(coef + (size_t)rowBase * kE))[tid];

  v8f acc[4][2];
#pragma unroll
  for (int t = 0; t < 4; ++t)
#pragma unroll
    for (int u = 0; u < 2; ++u) acc[t][u] = v8f{};

  const int ak0 = lhalf * 16, bk0 = lhalf * 16;

  for (int kk = 0; kk < KSTEPS; ++kk) {
    const int e = kk / STEPS_PER_EXPERT;
    const int kin = (kk % STEPS_PER_EXPERT) * BK;
    const float* gA = x + (size_t)rowBase * kIn + kin;
    const float* gB = w + ((size_t)e * kOut + colBase) * kIn + kin;
#pragma unroll
    for (int j = 0; j < 4; ++j) {
      const int c = tid + j * 256, r = c >> 3, q = (c & 7) * 4;
      *(v4f*)&lA[r * LDA + q] = *(const v4f*)(gA + (size_t)r * kIn + q);
      *(v4f*)&lB[r * LDB + q] = *(const v4f*)(gB + (size_t)r * kIn + q);
    }
    __syncthreads();
    v16bf bhi[2], blo[2];
#pragma unroll
    for (int u = 0; u < 2; ++u) {
      const float* p = &lB[(wn * 32 + u * 16 + l16) * LDB + bk0];
      float f[16] __attribute__((aligned(16)));
      *(v4f*)(f + 0) = *(const v4f*)(p + 0);  *(v4f*)(f + 4)  = *(const v4f*)(p + 4);
      *(v4f*)(f + 8) = *(const v4f*)(p + 8);  *(v4f*)(f + 12) = *(const v4f*)(p + 12);
      moe_split16(f, bhi[u], blo[u]);
    }
#pragma unroll
    for (int t = 0; t < 4; ++t) {
      const int arow = wm * 64 + t * 16 + l16;
      const float cf = lC[arow * kE + e];
      const float* p = &lA[arow * LDA + ak0];
      float f[16] __attribute__((aligned(16)));
      *(v4f*)(f + 0) = *(const v4f*)(p + 0);   *(v4f*)(f + 4)  = *(const v4f*)(p + 4);
      *(v4f*)(f + 8) = *(const v4f*)(p + 8);  *(v4f*)(f + 12) = *(const v4f*)(p + 12);
#pragma unroll
      for (int i = 0; i < 16; ++i) f[i] *= cf;
      v16bf ahi, alo;
      moe_split16(f, ahi, alo);
#pragma unroll
      for (int u = 0; u < 2; ++u) {
        acc[t][u] = __builtin_amdgcn_wmma_f32_16x16x32_bf16(false, ahi, false, bhi[u], (short)0, acc[t][u], false, false);
        acc[t][u] = __builtin_amdgcn_wmma_f32_16x16x32_bf16(false, ahi, false, blo[u], (short)0, acc[t][u], false, false);
        acc[t][u] = __builtin_amdgcn_wmma_f32_16x16x32_bf16(false, alo, false, bhi[u], (short)0, acc[t][u], false, false);
      }
    }
    __syncthreads();
  }
#pragma unroll
  for (int u = 0; u < 2; ++u) {
    const int col = colBase + wn * 32 + u * 16 + l16;
    float bc[8];
#pragma unroll
    for (int q = 0; q < 8; ++q) bc[q] = bias[q * kOut + col];
#pragma unroll
    for (int t = 0; t < 4; ++t)
#pragma unroll
      for (int j = 0; j < 8; ++j) {
        const int lrow = wm * 64 + t * 16 + lhalf * 8 + j;
        float bd = 0.f;
#pragma unroll
        for (int q = 0; q < 8; ++q) bd += lC[lrow * kE + q] * bc[q];
        out[(size_t)(rowBase + lrow) * kOut + col] = acc[t][u][j] + bd;
      }
  }
}

extern "C" void kernel_launch(void* const* d_in, const int* in_sizes, int n_in,
                              void* d_out, int out_size, void* d_ws, size_t ws_size,
                              hipStream_t stream) {
  const float* x    = (const float*)d_in[0];
  const float* coef = (const float*)d_in[1];
  const float* w    = (const float*)d_in[2];
  const float* b    = (const float*)d_in[3];
  float* out        = (float*)d_out;

  dim3 grid(kOut / BN, kN / BM);   // 4 x 128 workgroups
  const size_t need = kWElems * 2 * sizeof(__bf16);  // 8 MB hi+lo

  if (ws_size >= need) {
    __bf16* whi = (__bf16*)d_ws;
    __bf16* wlo = whi + kWElems;
    moe_prepack_w<<<dim3((unsigned)(kWElems / 4 / 256)), dim3(256), 0, stream>>>(w, whi, wlo);
    moe_gemm_main<<<grid, dim3(256), 0, stream>>>(x, coef, whi, wlo, b, out);
  } else {
    moe_gemm_fallback<<<grid, dim3(256), 0, stream>>>(x, coef, w, b, out);
  }
  (void)in_sizes; (void)n_in; (void)out_size;
}
